// AttentiveTransformer_22643067585022
// MI455X (gfx1250) — matrix operand order repaired and checked by reading, not compiled, not run on hardware
//
#include <hip/hip_runtime.h>
#include <hip/hip_bf16.h>

typedef __attribute__((ext_vector_type(16))) __bf16 v16bf;
typedef __attribute__((ext_vector_type(8)))  __bf16 v8bf;
typedef __attribute__((ext_vector_type(4)))  __bf16 v4bf;
typedef __attribute__((ext_vector_type(8)))  float  v8f;
typedef __attribute__((ext_vector_type(4)))  int    v4i;

#define DIN   1024
#define DOUT  1024
#define BROWS 65536
#define VBSZ  128

// GEMM tiling: block tile 128x128, K-step 64, 8 wave32 per block, wave tile 32x64
#define MT  128
#define NT  128
#define KT  64
#define LDK 72    // bf16 elems per A-tile row (64 + 8 pad)  -> 144 B (16B aligned)
#define LDN 136   // bf16 elems per B-tile row (128 + 8 pad) -> 272 B (16B aligned)

#define AS1 __attribute__((address_space(1)))
#define AS3 __attribute__((address_space(3)))

#if __has_builtin(__builtin_amdgcn_global_load_async_to_lds_b128)
#define HAVE_ASYNC_LDS 1
#else
#define HAVE_ASYNC_LDS 0
#endif

static __device__ __forceinline__ void wait_async0() {
#if __has_builtin(__builtin_amdgcn_s_wait_asynccnt)
  __builtin_amdgcn_s_wait_asynccnt(0);
#else
  asm volatile("s_wait_asynccnt 0x0" ::: "memory");
#endif
}

// ---------------------------------------------------------------------------
// Kernel 0a: elementwise f32 -> bf16 (feat), 4 elems/thread
// ---------------------------------------------------------------------------
__global__ __launch_bounds__(256) void convert_bf16(
    const float* __restrict__ in, __bf16* __restrict__ outp)
{
  size_t i = ((size_t)blockIdx.x * 256 + threadIdx.x) * 4;
  float4 v = *(const float4*)(in + i);
  v4bf o = { (__bf16)v.x, (__bf16)v.y, (__bf16)v.z, (__bf16)v.w };
  *(v4bf*)(outp + i) = o;
}

// ---------------------------------------------------------------------------
// Kernel 0b: Wt[k][n] = (bf16) W[n][k]  (LDS 32x32 tiles)
// ---------------------------------------------------------------------------
__global__ __launch_bounds__(256) void transpose_w_bf16(
    const float* __restrict__ W, __bf16* __restrict__ Wt)
{
  __shared__ float tile[32][33];
  const int kb = blockIdx.x * 32;       // k base (input minor dim)
  const int nb = blockIdx.y * 32;       // n base (input major dim)
  const int tx = threadIdx.x & 31;
  const int ty = threadIdx.x >> 5;      // 0..7
  #pragma unroll
  for (int i = 0; i < 32; i += 8)
    tile[ty + i][tx] = W[(size_t)(nb + ty + i) * DIN + kb + tx];
  __syncthreads();
  #pragma unroll
  for (int i = 0; i < 32; i += 8)
    Wt[(size_t)(kb + ty + i) * DOUT + nb + tx] = (__bf16)tile[tx][ty + i];
}

// ---------------------------------------------------------------------------
// Kernel 1: x = Abf @ Wt  via v_wmma_f32_16x16x32_bf16, fused GBN stats.
// M-tile (128) == virtual batch, so each block owns a full (vb, 128-col) slab
// and writes mean/rstd directly from its accumulators.
// ---------------------------------------------------------------------------
__global__ __launch_bounds__(256) void gemm_bf16_wmma(
    const __bf16* __restrict__ Abf, const __bf16* __restrict__ Wt,
    float* __restrict__ x, float* __restrict__ mean, float* __restrict__ rstd)
{
  __shared__ __bf16 Asm[MT * LDK];      // [m][k]   18432 B
  __shared__ __bf16 Bsm[KT * LDN];      // [k][n]   17408 B
  __shared__ float  sColS[4][NT];       // per-wm column partial sums
  __shared__ float  sColQ[4][NT];

  const int tid  = threadIdx.x;
  const int lane = tid & 31;
  const int wid  = tid >> 5;
  const int wm   = wid >> 1;            // 0..3 -> wave M offset wm*32
  const int wn   = wid & 1;             // 0..1 -> wave N offset wn*64
  const int half = lane >> 4;
  const int l16  = lane & 15;

  const int m0 = blockIdx.y * MT;       // == vb * VBSZ
  const int n0 = blockIdx.x * NT;

  v8f acc[2][4];
  #pragma unroll
  for (int i = 0; i < 2; ++i)
    #pragma unroll
    for (int j = 0; j < 4; ++j) { v8f z = {}; acc[i][j] = z; }

  // staging maps (pure b128 copies; bf16 already)
  const int ar  = tid >> 1;             // 0..127 A row
  const int asg = tid & 1;              //  segment (32 elems = 64 B)
  const int br  = tid >> 2;             // 0..63  B row (k)
  const int bsg = tid & 3;              //  segment (32 elems = 64 B)

  for (int k0 = 0; k0 < DIN; k0 += KT) {
    const v8bf* gA = (const v8bf*)(Abf + (size_t)(m0 + ar) * DIN + k0 + asg * 32);
    v8bf*       lA = (v8bf*)&Asm[ar * LDK + asg * 32];
    const v8bf* gB = (const v8bf*)(Wt + (size_t)(k0 + br) * DOUT + n0 + bsg * 32);
    v8bf*       lB = (v8bf*)&Bsm[br * LDN + bsg * 32];
#if HAVE_ASYNC_LDS
    #pragma unroll
    for (int i = 0; i < 4; ++i)
      __builtin_amdgcn_global_load_async_to_lds_b128(
          (AS1 v4i*)(gA + i), (AS3 v4i*)(lA + i), 0, 0);
    #pragma unroll
    for (int i = 0; i < 4; ++i)
      __builtin_amdgcn_global_load_async_to_lds_b128(
          (AS1 v4i*)(gB + i), (AS3 v4i*)(lB + i), 0, 0);
    wait_async0();
#else
    if (k0 + KT < DIN) {
      __builtin_prefetch(gA + 4 * (DIN / 8) /* 4 rows down, same K range */, 0, 1);
      __builtin_prefetch(Wt + (size_t)(k0 + KT + br) * DOUT + n0 + bsg * 32, 0, 1);
    }
    #pragma unroll
    for (int i = 0; i < 4; ++i) lA[i] = gA[i];
    #pragma unroll
    for (int i = 0; i < 4; ++i) lB[i] = gB[i];
#endif
    __syncthreads();

    union F { v16bf v; v8bf h[2]; };
    #pragma unroll
    for (int kc = 0; kc < 2; ++kc) {              // two K=32 chunks
      F a[2], b[4];
      #pragma unroll
      for (int sm = 0; sm < 2; ++sm) {
        const v8bf* arp = (const v8bf*)&Asm[(wm * 32 + sm * 16 + l16) * LDK];
        a[sm].h[0] = arp[kc * 4 + half];          // K = kc*32 + half*8 ..
        a[sm].h[1] = arp[kc * 4 + 2 + half];      // K = kc*32 + 16 + half*8 ..
      }
      #pragma unroll
      for (int sn = 0; sn < 4; ++sn) {
        const __bf16* brp = &Bsm[(kc * 32 + half * 8) * LDN + wn * 64 + sn * 16 + l16];
        #pragma unroll
        for (int s = 0; s < 8; ++s) {
          b[sn].v[s]     = brp[s * LDN];
          b[sn].v[8 + s] = brp[(16 + s) * LDN];
        }
      }
      #pragma unroll
      for (int sm = 0; sm < 2; ++sm)
        #pragma unroll
        for (int sn = 0; sn < 4; ++sn)
          acc[sm][sn] = __builtin_amdgcn_wmma_f32_16x16x32_bf16(
              false, a[sm].v, false, b[sn].v, (short)0, acc[sm][sn], false, false);
    }
    __syncthreads();
  }

  // ---- store x ----
  #pragma unroll
  for (int sm = 0; sm < 2; ++sm)
    #pragma unroll
    for (int sn = 0; sn < 4; ++sn)
      #pragma unroll
      for (int r = 0; r < 8; ++r) {
        int m = m0 + wm * 32 + sm * 16 + r + 8 * half;
        int n = n0 + wn * 64 + sn * 16 + l16;
        x[(size_t)m * DOUT + n] = acc[sm][sn][r];
      }

  // ---- fused GhostBatchNorm stats: this block owns all 128 rows of vb ----
  #pragma unroll
  for (int sn = 0; sn < 4; ++sn) {
    float cs = 0.f, cq = 0.f;
    #pragma unroll
    for (int sm = 0; sm < 2; ++sm)
      #pragma unroll
      for (int r = 0; r < 8; ++r) { float v = acc[sm][sn][r]; cs += v; cq += v * v; }
    cs += __shfl_xor(cs, 16, 32);                 // partner lane = other row-half
    cq += __shfl_xor(cq, 16, 32);
    if (half == 0) {
      int c = wn * 64 + sn * 16 + l16;
      sColS[wm][c] = cs;
      sColQ[wm][c] = cq;
    }
  }
  __syncthreads();
  if (tid < NT) {
    float S = sColS[0][tid] + sColS[1][tid] + sColS[2][tid] + sColS[3][tid];
    float Q = sColQ[0][tid] + sColQ[1][tid] + sColQ[2][tid] + sColQ[3][tid];
    float m = S * (1.0f / VBSZ);
    float var = Q * (1.0f / VBSZ) - m * m;        // biased var (matches reference)
    size_t idx = (size_t)blockIdx.y * DOUT + n0 + tid;
    mean[idx] = m;
    rstd[idx] = rsqrtf(var + 1e-5f);
  }
}

// ---------------------------------------------------------------------------
// Kernel 2: normalize * gamma + beta, * priors, exact sparsemax (Michelot)
// ---------------------------------------------------------------------------
__global__ __launch_bounds__(256) void norm_sparsemax(
    const float* __restrict__ x, const float* __restrict__ mean,
    const float* __restrict__ rstd, const float* __restrict__ gamma,
    const float* __restrict__ beta, const float* __restrict__ priors,
    float* __restrict__ out)
{
  __shared__ float sS[8];
  __shared__ float sC[8];
  const int row  = blockIdx.x;
  const int vb   = row >> 7;
  const int tid  = threadIdx.x;
  const int lane = tid & 31;
  const int wid  = tid >> 5;

  float v[4];
  #pragma unroll
  for (int j = 0; j < 4; ++j) {
    int c = tid + 256 * j;
    float xv = x[(size_t)row * DOUT + c];
    float xn = (xv - mean[vb * DOUT + c]) * rstd[vb * DOUT + c];
    v[j] = (xn * gamma[c] + beta[c]) * priors[(size_t)row * DOUT + c];
  }

  // Michelot: tau <- (sum_{v>tau} v - 1)/count until fixed point (exact projection)
  float tau = -3.0e38f;
  for (int it = 0; it < 64; ++it) {
    float s = 0.f, c = 0.f;
    #pragma unroll
    for (int j = 0; j < 4; ++j)
      if (v[j] > tau) { s += v[j]; c += 1.0f; }
    #pragma unroll
    for (int off = 16; off > 0; off >>= 1) {
      s += __shfl_xor(s, off, 32);
      c += __shfl_xor(c, off, 32);
    }
    __syncthreads();
    if (lane == 0) { sS[wid] = s; sC[wid] = c; }
    __syncthreads();
    float S = 0.f, C = 0.f;
    #pragma unroll
    for (int w = 0; w < 8; ++w) { S += sS[w]; C += sC[w]; }
    float tnew = (S - 1.0f) / C;
    if (tnew == tau) break;                       // uniform branch
    tau = tnew;
  }

  #pragma unroll
  for (int j = 0; j < 4; ++j) {
    int c = tid + 256 * j;
    out[(size_t)row * DOUT + c] = fmaxf(v[j] - tau, 0.0f);
  }
}

// ---------------------------------------------------------------------------
extern "C" void kernel_launch(void* const* d_in, const int* in_sizes, int n_in,
                              void* d_out, int out_size, void* d_ws, size_t ws_size,
                              hipStream_t stream) {
  const float* priors = (const float*)d_in[0];
  const float* feat   = (const float*)d_in[1];
  const float* W      = (const float*)d_in[2];
  const float* gamma  = (const float*)d_in[3];
  const float* beta   = (const float*)d_in[4];
  float* out = (float*)d_out;

  // workspace: x f32 (256MB) | Abf bf16 (128MB) | Wt bf16 (2MB) | mean | rstd
  char* ws = (char*)d_ws;
  float*  x    = (float*)ws;
  __bf16* Abf  = (__bf16*)(ws + (size_t)BROWS * DOUT * 4);
  __bf16* Wt   = (__bf16*)(ws + (size_t)BROWS * DOUT * 4 + (size_t)BROWS * DIN * 2);
  float*  mean = (float*)((char*)Wt + (size_t)DIN * DOUT * 2);
  float*  rstd = mean + (BROWS / VBSZ) * DOUT;

  convert_bf16<<<(BROWS * DIN) / (256 * 4), 256, 0, stream>>>(feat, Abf);  // 65536 blocks

  dim3 gW(DIN / 32, DOUT / 32);         // (32, 32)
  transpose_w_bf16<<<gW, 256, 0, stream>>>(W, Wt);

  dim3 g1(DOUT / NT, BROWS / MT);       // (8, 512)
  gemm_bf16_wmma<<<g1, 256, 0, stream>>>(Abf, Wt, x, mean, rstd);

  norm_sparsemax<<<BROWS, 256, 0, stream>>>(x, mean, rstd, gamma, beta, priors, out);
}
